// FeatureExtractorLayer_3865470567207
// MI455X (gfx1250) — matrix operand order repaired and checked by reading, not compiled, not run on hardware
//
#include <hip/hip_runtime.h>

typedef __attribute__((ext_vector_type(16))) __bf16        v16bf;
typedef __attribute__((ext_vector_type(8)))  float         v8f;
typedef __attribute__((ext_vector_type(4)))  unsigned int  v4u;

union Frag16 { v4u q[2]; v16bf v; };

__device__ __forceinline__ unsigned short f2bf(float f) {
  unsigned u = __float_as_uint(f);
  u += 0x7FFFu + ((u >> 16) & 1u);
  return (unsigned short)(u >> 16);
}

// order-preserving float->uint key (monotone), key 0 == most negative
__device__ __forceinline__ unsigned fkey(float f) {
  unsigned b = __float_as_uint(f);
  return (b & 0x80000000u) ? ~b : (b | 0x80000000u);
}
__device__ __forceinline__ float funkey(unsigned k) {
  unsigned b = (k & 0x80000000u) ? (k & 0x7FFFFFFFu) : ~k;
  return __uint_as_float(b);
}

// A fragment (16x32 bf16, row-major src): lane half h gets K [k+8h, k+8h+8) and [k+16+8h, ...)
__device__ __forceinline__ v16bf load_a(const unsigned short* row, int k, int half) {
  const unsigned short* p = row + k + half * 8;
  Frag16 f;
  f.q[0] = *(const v4u*)(p);
  f.q[1] = *(const v4u*)(p + 16);
  return f.v;
}
// B fragment (32x16 bf16), weights stored transposed [N, Kp] row-major:
// lane n = column N, half h gets K [k+8h, k+8h+8) and [k+16+8h, ...)
__device__ __forceinline__ v16bf load_b(const unsigned short* row, int k, int half) {
  const unsigned short* p = row + k + half * 8;
  Frag16 f;
  f.q[0] = *(const v4u*)(p);
  f.q[1] = *(const v4u*)(p + 16);
  return f.v;
}

// ---------------- weight transpose+convert:  W[K,128] f32 -> Wt[128, ldt] bf16 at col offset koff
__global__ void k_wtrans(const float* __restrict__ W, int K, unsigned short* __restrict__ Wt,
                         int ldt, int koff) {
  int i = blockIdx.x * blockDim.x + threadIdx.x;
  if (i >= K * 128) return;
  int k = i >> 7;        // i / 128
  int n = i & 127;       // i % 128
  Wt[(size_t)n * ldt + koff + k] = f2bf(W[(size_t)k * 128 + n]);
}

// ---------------- concat [a(da) | b(db) | zeros] f32 -> bf16, row width ld
__global__ void k_concat2(const float* __restrict__ a, int da,
                          const float* __restrict__ b, int db,
                          unsigned short* __restrict__ out, int ld, int n) {
  long long i = (long long)blockIdx.x * blockDim.x + threadIdx.x;
  long long total = (long long)n * ld;
  if (i >= total) return;
  int row = (int)(i / ld);
  int col = (int)(i % ld);
  float v = 0.f;
  if (col < da) v = a[(size_t)row * da + col];
  else if (col < da + db) v = b[(size_t)row * db + (col - da)];
  out[i] = f2bf(v);
}

// ---------------- build edge_comb [learned(128)|lo|hi|mm|wo(5)|zeros] -> bf16 ld=160
__global__ void k_build_edge(const float* __restrict__ learned, const float* __restrict__ lo,
                             const float* __restrict__ hi, const float* __restrict__ mm,
                             const float* __restrict__ wo, unsigned short* __restrict__ out,
                             int ne) {
  long long i = (long long)blockIdx.x * blockDim.x + threadIdx.x;
  long long total = (long long)ne * 160;
  if (i >= total) return;
  int e = (int)(i / 160);
  int c = (int)(i % 160);
  float v = 0.f;
  if (c < 128) v = learned[(size_t)e * 128 + c];
  else if (c == 128) v = lo[e];
  else if (c == 129) v = hi[e];
  else if (c == 130) v = mm[e];
  else if (c < 136)  v = wo[(size_t)e * 5 + (c - 131)];
  out[i] = f2bf(v);
}

// ---------------- WMMA GEMM: C[M,128] = act(A[M,Kp]_bf16 @ Wt^T + bias)
// block = 128 threads = 4 waves; block computes 16 rows x 128 cols; wave w -> col tiles 2w, 2w+1
__global__ void k_gemm128(const unsigned short* __restrict__ A, int lda,
                          const unsigned short* __restrict__ Wt,
                          const float* __restrict__ bias,
                          float* __restrict__ outF, unsigned short* __restrict__ outB,
                          int M, int Kp, int relu) {
  int wave = threadIdx.x >> 5;
  int lane = threadIdx.x & 31;
  int half = lane >> 4;
  int r    = lane & 15;
  int m0   = blockIdx.x * 16;
  int arow = m0 + r; if (arow >= M) arow = M - 1;
  const unsigned short* Arow = A + (size_t)arow * lda;
  int n0 = wave * 32;
  const unsigned short* W0 = Wt + (size_t)(n0 + r) * Kp;
  const unsigned short* W1 = Wt + (size_t)(n0 + 16 + r) * Kp;
  v8f c0 = {};
  v8f c1 = {};
  for (int k = 0; k < Kp; k += 32) {
    v16bf a  = load_a(Arow, k, half);
    v16bf b0 = load_b(W0, k, half);
    v16bf b1 = load_b(W1, k, half);
    c0 = __builtin_amdgcn_wmma_f32_16x16x32_bf16(false, a, false, b0, (short)0, c0, false, false);
    c1 = __builtin_amdgcn_wmma_f32_16x16x32_bf16(false, a, false, b1, (short)0, c1, false, false);
  }
  int mb = m0 + 8 * half;
  for (int vg = 0; vg < 8; vg++) {
    int m = mb + vg;
    if (m < M) {
      int n = n0 + r;
      float x = c0[vg] + (bias ? bias[n] : 0.f);
      if (relu) x = fmaxf(x, 0.f);
      if (outF) outF[(size_t)m * 128 + n] = x;
      if (outB) outB[(size_t)m * 128 + n] = f2bf(x);
      n += 16;
      float y = c1[vg] + (bias ? bias[n] : 0.f);
      if (relu) y = fmaxf(y, 0.f);
      if (outF) outF[(size_t)m * 128 + n] = y;
      if (outB) outB[(size_t)m * 128 + n] = f2bf(y);
    }
  }
}

// ---------------- gather-fused edge-MLP layer 1:
// t1[e,:] = relu( [edge_comb(e) | vc[ev[e]] | cc[ec[e]]] @ We1^T + b1 ),  K = 160+128+128 = 416
__global__ void k_gemm_edge_h(const unsigned short* __restrict__ Ecomb,   // ld 160
                              const unsigned short* __restrict__ Vc,      // ld 128
                              const unsigned short* __restrict__ Cc,      // ld 128
                              const int* __restrict__ ev, const int* __restrict__ ec,
                              const unsigned short* __restrict__ Wt,      // [128, 416]
                              const float* __restrict__ bias,
                              unsigned short* __restrict__ outB, int M) {
  const int Kp = 416;
  int wave = threadIdx.x >> 5;
  int lane = threadIdx.x & 31;
  int half = lane >> 4;
  int r    = lane & 15;
  int m0   = blockIdx.x * 16;
  int e = m0 + r; if (e >= M) e = M - 1;
  const unsigned short* pe = Ecomb + (size_t)e * 160;
  const unsigned short* pv = Vc + (size_t)ev[e] * 128;
  const unsigned short* pc = Cc + (size_t)ec[e] * 128;
  int n0 = wave * 32;
  const unsigned short* W0 = Wt + (size_t)(n0 + r) * Kp;
  const unsigned short* W1 = Wt + (size_t)(n0 + 16 + r) * Kp;
  v8f c0 = {};
  v8f c1 = {};
  for (int k = 0; k < Kp; k += 32) {
    const unsigned short* src; int kk;
    if (k < 160)      { src = pe; kk = k; }
    else if (k < 288) { src = pv; kk = k - 160; }
    else              { src = pc; kk = k - 288; }
    v16bf a  = load_a(src, kk, half);
    v16bf b0 = load_b(W0, k, half);
    v16bf b1 = load_b(W1, k, half);
    c0 = __builtin_amdgcn_wmma_f32_16x16x32_bf16(false, a, false, b0, (short)0, c0, false, false);
    c1 = __builtin_amdgcn_wmma_f32_16x16x32_bf16(false, a, false, b1, (short)0, c1, false, false);
  }
  int mb = m0 + 8 * half;
  for (int vg = 0; vg < 8; vg++) {
    int m = mb + vg;
    if (m < M) {
      int n = n0 + r;
      float x = fmaxf(c0[vg] + bias[n], 0.f);
      outB[(size_t)m * 128 + n] = f2bf(x);
      n += 16;
      float y = fmaxf(c1[vg] + bias[n], 0.f);
      outB[(size_t)m * 128 + n] = f2bf(y);
    }
  }
}

// ---------------- attention alpha: wave per edge, alpha = <q[dst], k[src]+ee[e]> / sqrt(128)
__global__ void k_alpha(const float* __restrict__ q, const float* __restrict__ kb,
                        const float* __restrict__ ee,
                        const int* __restrict__ dstI, const int* __restrict__ srcI,
                        float* __restrict__ alpha, unsigned* __restrict__ amax, int ne) {
  int wid  = (int)(((long long)blockIdx.x * blockDim.x + threadIdx.x) >> 5);
  int lane = threadIdx.x & 31;
  if (wid >= ne) return;
  int d = dstI[wid], s = srcI[wid];
  const float* qr = q  + (size_t)d * 128;
  const float* kr = kb + (size_t)s * 128;
  const float* er = ee + (size_t)wid * 128;
  float sum = 0.f;
  for (int c = lane; c < 128; c += 32) sum += qr[c] * (kr[c] + er[c]);
  sum += __shfl_xor(sum, 16);
  sum += __shfl_xor(sum, 8);
  sum += __shfl_xor(sum, 4);
  sum += __shfl_xor(sum, 2);
  sum += __shfl_xor(sum, 1);
  if (lane == 0) {
    float al = sum * 0.08838834764831845f;  // 1/sqrt(128)
    alpha[wid] = al;
    atomicMax(&amax[d], fkey(al));
  }
}

// ---------------- ex = exp(alpha - amax[dst]); denom += ex; cnt += 1   (alpha overwritten in place)
__global__ void k_exp(float* __restrict__ alpha, const unsigned* __restrict__ amax,
                      const int* __restrict__ dstI,
                      float* __restrict__ denom, float* __restrict__ cnt, int ne) {
  int e = (int)((long long)blockIdx.x * blockDim.x + threadIdx.x);
  if (e >= ne) return;
  int d = dstI[e];
  float ex = __expf(alpha[e] - funkey(amax[d]));
  alpha[e] = ex;
  atomicAdd(&denom[d], ex);
  atomicAdd(&cnt[d], 1.0f);
}

// ---------------- scatter weighted messages: agg[dst] += (v[src]+ee[e]) * ex/denom[dst]
__global__ void k_scatter(const float* __restrict__ vb, const float* __restrict__ ee,
                          const float* __restrict__ exb, const float* __restrict__ denom,
                          const int* __restrict__ dstI, const int* __restrict__ srcI,
                          float* __restrict__ agg, int ne) {
  int wid  = (int)(((long long)blockIdx.x * blockDim.x + threadIdx.x) >> 5);
  int lane = threadIdx.x & 31;
  if (wid >= ne) return;
  int d = dstI[wid], s = srcI[wid];
  float a = exb[wid] / denom[d];
  const float* vr = vb + (size_t)s * 128;
  const float* er = ee + (size_t)wid * 128;
  float* ar = agg + (size_t)d * 128;
  for (int c = lane; c < 128; c += 32) atomicAdd(&ar[c], (vr[c] + er[c]) * a);
}

// ---------------- out = relu(agg/max(cnt,1) + skip)
__global__ void k_finalize(const float* __restrict__ agg, const float* __restrict__ cnt,
                           const float* __restrict__ skip, float* __restrict__ out, int n) {
  long long i = (long long)blockIdx.x * blockDim.x + threadIdx.x;
  if (i >= (long long)n * 128) return;
  int node = (int)(i >> 7);
  float c = fmaxf(cnt[node], 1.0f);
  out[i] = fmaxf(agg[i] / c + skip[i], 0.0f);
}

extern "C" void kernel_launch(void* const* d_in, const int* in_sizes, int n_in,
                              void* d_out, int out_size, void* d_ws, size_t ws_size,
                              hipStream_t stream) {
  (void)n_in; (void)out_size; (void)ws_size;
  const int NV = in_sizes[0] / 128;
  const int NC = in_sizes[2] / 128;
  const int NE = in_sizes[4] / 128;
  const int KP = 160;          // padded K for 144/136-wide inputs
  const int KH = 416;          // padded K for 392-wide edge-MLP input

  const float* var_learned  = (const float*)d_in[0];
  const float* var_lp       = (const float*)d_in[1];
  const float* con_learned  = (const float*)d_in[2];
  const float* con_lp       = (const float*)d_in[3];
  const float* edge_learned = (const float*)d_in[4];
  const float* lo_costs     = (const float*)d_in[5];
  const float* hi_costs     = (const float*)d_in[6];
  const float* def_mm       = (const float*)d_in[7];
  const float* edge_wo      = (const float*)d_in[8];
  const float* cu_Wq = (const float*)d_in[9];  const float* cu_bq = (const float*)d_in[10];
  const float* cu_Wk = (const float*)d_in[11]; const float* cu_bk = (const float*)d_in[12];
  const float* cu_Wv = (const float*)d_in[13]; const float* cu_bv = (const float*)d_in[14];
  const float* cu_We = (const float*)d_in[15];
  const float* cu_Ws = (const float*)d_in[16]; const float* cu_bs = (const float*)d_in[17];
  const float* vu_Wq = (const float*)d_in[18]; const float* vu_bq = (const float*)d_in[19];
  const float* vu_Wk = (const float*)d_in[20]; const float* vu_bk = (const float*)d_in[21];
  const float* vu_Wv = (const float*)d_in[22]; const float* vu_bv = (const float*)d_in[23];
  const float* vu_We = (const float*)d_in[24];
  const float* vu_Ws = (const float*)d_in[25]; const float* vu_bs = (const float*)d_in[26];
  const float* eu_vW1 = (const float*)d_in[27]; const float* eu_vb1 = (const float*)d_in[28];
  const float* eu_vW2 = (const float*)d_in[29]; const float* eu_vb2 = (const float*)d_in[30];
  const float* eu_cW1 = (const float*)d_in[31]; const float* eu_cb1 = (const float*)d_in[32];
  const float* eu_cW2 = (const float*)d_in[33]; const float* eu_cb2 = (const float*)d_in[34];
  const float* eu_eW1 = (const float*)d_in[35]; const float* eu_eb1 = (const float*)d_in[36];
  const float* eu_eW2 = (const float*)d_in[37]; const float* eu_eb2 = (const float*)d_in[38];
  const int* eiv = (const int*)d_in[39];   // edge_index_var  (var endpoint of each edge)
  const int* eic = (const int*)d_in[40];   // edge_index_con

  // ---- workspace carve-out
  char* ws = (char*)d_ws;
  size_t off = 0;
  auto take = [&](size_t bytes) -> char* {
    char* p = ws + off;
    off = (off + bytes + 255) & ~(size_t)255;
    return p;
  };

  char* wt_start = ws + off;
  unsigned short* WcuWq = (unsigned short*)take((size_t)128 * KP * 2);
  unsigned short* WcuWk = (unsigned short*)take((size_t)128 * KP * 2);
  unsigned short* WcuWv = (unsigned short*)take((size_t)128 * KP * 2);
  unsigned short* WcuWe = (unsigned short*)take((size_t)128 * KP * 2);
  unsigned short* WcuWs = (unsigned short*)take((size_t)128 * KP * 2);
  unsigned short* WvuWq = (unsigned short*)take((size_t)128 * KP * 2);
  unsigned short* WvuWk = (unsigned short*)take((size_t)128 * KP * 2);
  unsigned short* WvuWv = (unsigned short*)take((size_t)128 * KP * 2);
  unsigned short* WvuWe = (unsigned short*)take((size_t)128 * KP * 2);
  unsigned short* WvuWs = (unsigned short*)take((size_t)128 * KP * 2);
  unsigned short* WeuvW1 = (unsigned short*)take((size_t)128 * KP * 2);
  unsigned short* WeuvW2 = (unsigned short*)take((size_t)128 * 128 * 2);
  unsigned short* WeucW1 = (unsigned short*)take((size_t)128 * KP * 2);
  unsigned short* WeucW2 = (unsigned short*)take((size_t)128 * 128 * 2);
  unsigned short* WeueW1 = (unsigned short*)take((size_t)128 * KH * 2);
  unsigned short* WeueW2 = (unsigned short*)take((size_t)128 * 128 * 2);
  size_t wt_bytes = (size_t)((ws + off) - wt_start);

  unsigned short* varb  = (unsigned short*)take((size_t)NV * KP * 2);
  unsigned short* conb  = (unsigned short*)take((size_t)NC * KP * 2);
  unsigned short* edgeb = (unsigned short*)take((size_t)NE * KP * 2);
  float* eef   = (float*)take((size_t)NE * 128 * 4);
  float* qf    = (float*)take((size_t)NV * 128 * 4);
  float* kf    = (float*)take((size_t)NV * 128 * 4);
  float* vf    = (float*)take((size_t)NV * 128 * 4);
  float* skipf = (float*)take((size_t)NV * 128 * 4);
  float* aggf  = (float*)take((size_t)NV * 128 * 4);
  float* alphaf = (float*)take((size_t)NE * 4);
  unsigned* amax = (unsigned*)take((size_t)NV * 4);
  float* denom   = (float*)take((size_t)NV * 4);
  float* cntf    = (float*)take((size_t)NV * 4);
  unsigned short* vcb = (unsigned short*)take((size_t)NV * 128 * 2);
  unsigned short* ccb = (unsigned short*)take((size_t)NC * 128 * 2);
  unsigned short* nhb = (unsigned short*)take((size_t)NV * 128 * 2);
  unsigned short* t1b = (unsigned short*)take((size_t)NE * 128 * 2);

  float* out_var  = (float*)d_out;
  float* out_con  = out_var + (size_t)NV * 128;
  float* out_edge = out_con + (size_t)NC * 128;

  // ---- weight prep (zero pad regions, then transpose-fill)
  hipMemsetAsync(wt_start, 0, wt_bytes, stream);
  auto wtrans = [&](const float* W, int K, unsigned short* Wt, int ldt, int koff) {
    int tot = K * 128;
    k_wtrans<<<(tot + 255) / 256, 256, 0, stream>>>(W, K, Wt, ldt, koff);
  };
  wtrans(cu_Wq, 144, WcuWq, KP, 0);
  wtrans(cu_Wk, 144, WcuWk, KP, 0);
  wtrans(cu_Wv, 144, WcuWv, KP, 0);
  wtrans(cu_We, 136, WcuWe, KP, 0);
  wtrans(cu_Ws, 144, WcuWs, KP, 0);
  wtrans(vu_Wq, 144, WvuWq, KP, 0);
  wtrans(vu_Wk, 144, WvuWk, KP, 0);
  wtrans(vu_Wv, 144, WvuWv, KP, 0);
  wtrans(vu_We, 136, WvuWe, KP, 0);
  wtrans(vu_Ws, 144, WvuWs, KP, 0);
  wtrans(eu_vW1, 144, WeuvW1, KP, 0);
  wtrans(eu_vW2, 128, WeuvW2, 128, 0);
  wtrans(eu_cW1, 144, WeucW1, KP, 0);
  wtrans(eu_cW2, 128, WeucW2, 128, 0);
  wtrans(eu_eW1,              136, WeueW1, KH, 0);     // edge_comb segment
  wtrans(eu_eW1 + 136 * 128,  128, WeueW1, KH, 160);   // vc segment
  wtrans(eu_eW1 + 264 * 128,  128, WeueW1, KH, 288);   // cc segment
  wtrans(eu_eW2, 128, WeueW2, 128, 0);

  // ---- activation prep
  {
    long long t;
    t = (long long)NV * KP;
    k_concat2<<<(unsigned)((t + 255) / 256), 256, 0, stream>>>(var_learned, 128, var_lp, 16, varb, KP, NV);
    t = (long long)NC * KP;
    k_concat2<<<(unsigned)((t + 255) / 256), 256, 0, stream>>>(con_learned, 128, con_lp, 16, conb, KP, NC);
    t = (long long)NE * 160;
    k_build_edge<<<(unsigned)((t + 255) / 256), 256, 0, stream>>>(edge_learned, lo_costs, hi_costs,
                                                                  def_mm, edge_wo, edgeb, NE);
  }

  auto gemm = [&](const unsigned short* A, int lda, const unsigned short* Wt, int Kp,
                  const float* bias, float* oF, unsigned short* oB, int M, int relu) {
    k_gemm128<<<(M + 15) / 16, 128, 0, stream>>>(A, lda, Wt, bias, oF, oB, M, Kp, relu);
  };

  auto conv = [&](const unsigned short* xsrc, int Msrc, const unsigned short* xdst, int Mdst,
                  const unsigned short* WtQ, const float* bq2, const unsigned short* WtK,
                  const float* bk2, const unsigned short* WtV, const float* bv2,
                  const unsigned short* WtE, const unsigned short* WtS, const float* bs2,
                  const int* srcI, const int* dstI, float* outNode) {
    gemm(xdst, KP, WtQ, KP, bq2, qf, nullptr, Mdst, 0);
    gemm(xsrc, KP, WtK, KP, bk2, kf, nullptr, Msrc, 0);
    gemm(xsrc, KP, WtV, KP, bv2, vf, nullptr, Msrc, 0);
    gemm(edgeb, KP, WtE, KP, nullptr, eef, nullptr, NE, 0);
    gemm(xdst, KP, WtS, KP, bs2, skipf, nullptr, Mdst, 0);
    hipMemsetAsync(amax, 0, (size_t)Mdst * 4, stream);
    hipMemsetAsync(denom, 0, (size_t)Mdst * 4, stream);
    hipMemsetAsync(cntf, 0, (size_t)Mdst * 4, stream);
    hipMemsetAsync(aggf, 0, (size_t)Mdst * 128 * 4, stream);
    k_alpha<<<(NE + 7) / 8, 256, 0, stream>>>(qf, kf, eef, dstI, srcI, alphaf, amax, NE);
    k_exp<<<(NE + 255) / 256, 256, 0, stream>>>(alphaf, amax, dstI, denom, cntf, NE);
    k_scatter<<<(NE + 7) / 8, 256, 0, stream>>>(vf, eef, alphaf, denom, dstI, srcI, aggf, NE);
    long long t = (long long)Mdst * 128;
    k_finalize<<<(unsigned)((t + 255) / 256), 256, 0, stream>>>(aggf, cntf, skipf, outNode, Mdst);
  };

  // ---- con update: messages var -> con
  conv(varb, NV, conb, NC, WcuWq, cu_bq, WcuWk, cu_bk, WcuWv, cu_bv, WcuWe, WcuWs, cu_bs,
       eiv, eic, out_con);
  {
    long long t = (long long)NC * KP;   // rebuild con_comb from con_new
    k_concat2<<<(unsigned)((t + 255) / 256), 256, 0, stream>>>(out_con, 128, con_lp, 16, conb, KP, NC);
  }

  // ---- var update: messages con -> var
  conv(conb, NC, varb, NV, WvuWq, vu_bq, WvuWk, vu_bk, WvuWv, vu_bv, WvuWe, WvuWs, vu_bs,
       eic, eiv, out_var);
  {
    long long t = (long long)NV * KP;   // rebuild var_comb from var_new
    k_concat2<<<(unsigned)((t + 255) / 256), 256, 0, stream>>>(out_var, 128, var_lp, 16, varb, KP, NV);
  }

  // ---- edge update
  gemm(varb, KP, WeuvW1, KP, eu_vb1, nullptr, nhb, NV, 1);
  gemm(nhb, 128, WeuvW2, 128, eu_vb2, nullptr, vcb, NV, 1);
  gemm(conb, KP, WeucW1, KP, eu_cb1, nullptr, nhb, NC, 1);
  gemm(nhb, 128, WeucW2, 128, eu_cb2, nullptr, ccb, NC, 1);
  k_gemm_edge_h<<<(NE + 15) / 16, 128, 0, stream>>>(edgeb, vcb, ccb, eiv, eic, WeueW1, eu_eb1, t1b, NE);
  gemm(t1b, 128, WeueW2, 128, eu_eb2, out_edge, nullptr, NE, 1);
}
